// TobitKalmanLSTM_65566970740892
// MI455X (gfx1250) — matrix operand order repaired and checked by reading, not compiled, not run on hardware
//
#include <hip/hip_runtime.h>
#include <hip/hip_bf16.h>
#include <stdint.h>

// Problem constants (match reference)
#define T_MAX   128
#define B_SZ    256
#define OBS     32
#define HID     512
#define STATE   64
#define GATES   (4*HID)        // 2048
#define KCAT    (STATE + HID)  // 576
#define EPSF    1e-4f

typedef __attribute__((ext_vector_type(16))) _Float16 v16h;
typedef __attribute__((ext_vector_type(8)))  float    v8f;

#define WMMA_F16(a,b,c) \
  __builtin_amdgcn_wmma_f32_16x16x32_f16(false,(a),false,(b),(short)0,(c),false,false)

// ---------------- WMMA fragment helpers (wave32, 16x16x32 f16) ---------------
// A (16x32 MxK): lanes 0-15 -> M rows, lane>>4 -> K half.
__device__ __forceinline__ v16h frag_a(const _Float16* base, int lda, int m0,
                                       int k0, int lane) {
  const int m = m0 + (lane & 15);
  const int g = lane >> 4;
  const _Float16* p = base + m * lda + k0;
  v16h a;
#pragma unroll
  for (int i = 0; i < 8; ++i) a[i] = p[16*g + i];
#pragma unroll
  for (int i = 0; i < 8; ++i) a[8+i] = p[16*g + 8 + i];
  return a;
}
// B (32x16 KxN) stored N-major (Bst[n][k]): lanes 0-15 -> N, lane>>4 -> K half.
__device__ __forceinline__ v16h frag_b_nmajor(const _Float16* base, int ldb,
                                              int n0, int k0, int lane) {
  const int n = n0 + (lane & 15);
  const int g = lane >> 4;
  const _Float16* p = base + n * ldb + k0 + 16*g;
  v16h b;
#pragma unroll
  for (int i = 0; i < 16; ++i) b[i] = p[i];
  return b;
}
// B stored K-major (Bst[k][n]) — used when B is a plain row-major matrix.
__device__ __forceinline__ v16h frag_b_kmajor(const _Float16* base, int ldb,
                                              int n0, int k0, int lane) {
  const int n = n0 + (lane & 15);
  const int g = lane >> 4;
  v16h b;
#pragma unroll
  for (int i = 0; i < 16; ++i) b[i] = base[(k0 + 16*g + i) * ldb + n];
  return b;
}

__device__ __forceinline__ float sigmoidf_(float x) { return 1.f / (1.f + __expf(-x)); }
__device__ __forceinline__ float ncdf_(float z) { return 0.5f * (1.f + erff(z * 0.70710678118654752f)); }
__device__ __forceinline__ float npdf_(float z) { return 0.3989422804014327f * __expf(-0.5f * z * z); }

// ------------------------------ prep kernels --------------------------------
__global__ void pack_lstm_w_kernel(const float* __restrict__ wih,
                                   const float* __restrict__ whh,
                                   _Float16* __restrict__ wcat) {
  int idx = blockIdx.x * 256 + threadIdx.x;
  if (idx >= GATES * KCAT) return;
  int n = idx / KCAT, k = idx - n * KCAT;
  float v = (k < STATE) ? wih[n * STATE + k] : whh[n * HID + (k - STATE)];
  wcat[idx] = (_Float16)v;
}
__global__ void combine_bias_kernel(const float* __restrict__ bih,
                                    const float* __restrict__ bhh,
                                    float* __restrict__ bias) {
  int idx = blockIdx.x * 256 + threadIdx.x;
  if (idx < GATES) bias[idx] = bih[idx] + bhh[idx];
}
__global__ void cvt_f16_kernel(const float* __restrict__ src,
                               _Float16* __restrict__ dst, int n) {
  int idx = blockIdx.x * 256 + threadIdx.x;
  if (idx < n) dst[idx] = (_Float16)src[idx];
}
__global__ void bcast_y_kernel(const float* __restrict__ y0, float* __restrict__ yst) {
  int idx = blockIdx.x * 256 + threadIdx.x;
  if (idx < B_SZ * STATE) yst[idx] = y0[idx & (STATE - 1)];
}
__global__ void bcast_P_kernel(const float* __restrict__ P0, float* __restrict__ P) {
  int idx = blockIdx.x * 256 + threadIdx.x;
  if (idx < B_SZ * STATE * STATE) P[idx] = P0[idx & (STATE * STATE - 1)];
}
__global__ void zero_kernel(float* __restrict__ p, int n) {
  int idx = blockIdx.x * 256 + threadIdx.x;
  if (idx < n) p[idx] = 0.f;
}

// --------------------- Big GEMM: [B,ktot] x [ktot,Nout] ---------------------
// Serves LSTM gates (xin=[B,64] ++ hin=[B,512], ktot=576) and F_fc
// (xin=nullptr, ktot=512, Nout=4096). Block tile 64(M) x 128(N); 8 waves as
// 2x4, each wave 32x32 = 2x2 WMMA frags. Double-buffered LDS: global loads
// for slice k+1 are issued before the barrier and overlap the WMMAs of
// slice k, so each K-slice pays a single workgroup barrier.
__global__ __launch_bounds__(256) void big_gemm_kernel(
    const float* __restrict__ xin,     // optional [B, STATE] (may be nullptr)
    const float* __restrict__ hin,     // [B, HID]
    const _Float16* __restrict__ w,    // [Nout, ktot] f16 row-major
    const float* __restrict__ bias,    // optional [Nout]
    float* __restrict__ out,           // [B, Nout]
    int Nout, int ktot)
{
  __shared__ _Float16 As[2][64][32];
  __shared__ _Float16 Bs[2][128][32];
  const int tid = threadIdx.x, lane = tid & 31, wv = tid >> 5;
  const int M0 = blockIdx.y * 64, N0 = blockIdx.x * 128;
  const int wm = wv & 1, wn = wv >> 1;
  const int NK = ktot / 32;
  v8f acc[2][2] = {};

  float aR[8];     // staged A values (f32, converted on LDS store)
  uint4 bR[2];     // staged B rows (f16, 8 halves per uint4)
  const int rA = tid >> 2, kkA = (tid & 3) * 8;

  auto loadStage = [&](int kb) {
    const int k0 = kb * 32;
    const int bb = M0 + rA;
#pragma unroll
    for (int i = 0; i < 8; ++i) {
      const int k = k0 + kkA + i;
      if (xin)
        aR[i] = (k < STATE) ? xin[(size_t)bb * STATE + k]
                            : hin[(size_t)bb * HID + (k - STATE)];
      else
        aR[i] = hin[(size_t)bb * HID + k];
    }
#pragma unroll
    for (int c2 = 0; c2 < 2; ++c2) {
      const int idx = tid + c2 * 256;
      const int n = idx >> 2, kk = (idx & 3) * 8;
      bR[c2] = *reinterpret_cast<const uint4*>(w + (size_t)(N0 + n) * ktot + k0 + kk);
    }
    if (kb + 1 < NK)  // WGP-scope prefetch of next weight slice
      __builtin_prefetch(w + (size_t)(N0 + (tid >> 1)) * ktot + k0 + 32, 0, 3);
  };
  auto storeStage = [&](int buf) {
#pragma unroll
    for (int i = 0; i < 8; ++i) As[buf][rA][kkA + i] = (_Float16)aR[i];
#pragma unroll
    for (int c2 = 0; c2 < 2; ++c2) {
      const int idx = tid + c2 * 256;
      const int n = idx >> 2, kk = (idx & 3) * 8;
      *reinterpret_cast<uint4*>(&Bs[buf][n][kk]) = bR[c2];
    }
  };

  loadStage(0);
  storeStage(0);
  for (int kb = 0; kb < NK; ++kb) {
    const int cur = kb & 1;
    if (kb + 1 < NK) loadStage(kb + 1);
    __syncthreads();
    v16h a0 = frag_a(&As[cur][0][0], 32, wm * 32,      0, lane);
    v16h a1 = frag_a(&As[cur][0][0], 32, wm * 32 + 16, 0, lane);
    v16h b0 = frag_b_nmajor(&Bs[cur][0][0], 32, wn * 32,      0, lane);
    v16h b1 = frag_b_nmajor(&Bs[cur][0][0], 32, wn * 32 + 16, 0, lane);
    acc[0][0] = WMMA_F16(a0, b0, acc[0][0]);
    acc[0][1] = WMMA_F16(a0, b1, acc[0][1]);
    acc[1][0] = WMMA_F16(a1, b0, acc[1][0]);
    acc[1][1] = WMMA_F16(a1, b1, acc[1][1]);
    if (kb + 1 < NK) storeStage(1 - cur);
  }

  const int n = lane & 15, mg = (lane >> 4) * 8;
#pragma unroll
  for (int i = 0; i < 2; ++i)
#pragma unroll
    for (int j = 0; j < 2; ++j) {
      const int col = N0 + wn * 32 + j * 16 + n;
      const float bv = bias ? bias[col] : 0.f;
#pragma unroll
      for (int r = 0; r < 8; ++r) {
        const int row = M0 + wm * 32 + i * 16 + mg + r;
        out[(size_t)row * Nout + col] = acc[i][j][r] + bv;
      }
    }
}

// --------------------------- LSTM pointwise update --------------------------
__global__ __launch_bounds__(256) void lstm_pointwise_kernel(
    const float* __restrict__ gates, const float* __restrict__ bias,
    float* __restrict__ h, float* __restrict__ c) {
  int idx = blockIdx.x * 256 + threadIdx.x;
  if (idx >= B_SZ * HID) return;
  const int bb = idx >> 9, j = idx & (HID - 1);
  const float* g = gates + (size_t)bb * GATES;
  float i_ = sigmoidf_(g[j]           + bias[j]);
  float f_ = sigmoidf_(g[HID + j]     + bias[HID + j]);
  float g_ = tanhf    (g[2*HID + j]   + bias[2*HID + j]);
  float o_ = sigmoidf_(g[3*HID + j]   + bias[3*HID + j]);
  float cn = f_ * c[idx] + i_ * g_;
  c[idx] = cn;
  h[idx] = o_ * tanhf(cn);
}

// ---------------- Small FC GEMM: [B,512]x[512,N], N in {64,32} --------------
// Block tile 64(M) x 32(N), 8 waves as 4x2, one 16x16 frag per wave.
// Same double-buffered LDS pipeline as the big GEMM.
__global__ __launch_bounds__(256) void fc_gemm_kernel(
    const float* __restrict__ hin, const _Float16* __restrict__ w,
    const float* __restrict__ bias, float* __restrict__ out,
    float* __restrict__ out_dup, int N, int applyExp) {
  __shared__ _Float16 As[2][64][32];
  __shared__ _Float16 Bs[2][32][32];
  const int tid = threadIdx.x, lane = tid & 31, wv = tid >> 5;
  const int M0 = blockIdx.y * 64, N0 = blockIdx.x * 32;
  const int wm = wv >> 1, wn = wv & 1;
  const int NK = HID / 32;
  v8f acc = {};

  float aR[8];
  uint2 bR;
  const int rA = tid >> 2, kkA = (tid & 3) * 8;
  const int nB = tid >> 3, kkB = (tid & 7) * 4;

  auto loadStage = [&](int kb) {
    const int k0 = kb * 32;
    const float* src = hin + (size_t)(M0 + rA) * HID + k0 + kkA;
#pragma unroll
    for (int i = 0; i < 8; ++i) aR[i] = src[i];
    bR = *reinterpret_cast<const uint2*>(w + (size_t)(N0 + nB) * HID + k0 + kkB);
  };
  auto storeStage = [&](int buf) {
#pragma unroll
    for (int i = 0; i < 8; ++i) As[buf][rA][kkA + i] = (_Float16)aR[i];
    *reinterpret_cast<uint2*>(&Bs[buf][nB][kkB]) = bR;
  };

  loadStage(0);
  storeStage(0);
  for (int kb = 0; kb < NK; ++kb) {
    const int cur = kb & 1;
    if (kb + 1 < NK) loadStage(kb + 1);
    __syncthreads();
    v16h a = frag_a(&As[cur][0][0], 32, wm * 16, 0, lane);
    v16h b = frag_b_nmajor(&Bs[cur][0][0], 32, wn * 16, 0, lane);
    acc = WMMA_F16(a, b, acc);
    if (kb + 1 < NK) storeStage(1 - cur);
  }

  const int n = lane & 15, mg = (lane >> 4) * 8;
  const int col = N0 + wn * 16 + n;
  const float bv = bias[col];
#pragma unroll
  for (int r = 0; r < 8; ++r) {
    const int row = M0 + wm * 16 + mg + r;
    float v = acc[r] + bv;
    if (applyExp) v = __expf(v);
    out[(size_t)row * N + col] = v;
    if (out_dup) out_dup[(size_t)row * N + col] = v;
  }
}

// ------------------- Ppred = F P F^T + diag(Q), per batch -------------------
// One 128-thread block (4 waves) per batch element; two WMMA phases from LDS.
__global__ __launch_bounds__(128) void ppred_kernel(
    const float* __restrict__ F, const float* __restrict__ P,
    const float* __restrict__ Qd, float* __restrict__ Pp) {
  __shared__ _Float16 Fs[STATE][STATE];
  __shared__ _Float16 Ps[STATE][STATE];
  __shared__ _Float16 M1[STATE][STATE];
  const int b = blockIdx.x, tid = threadIdx.x, lane = tid & 31, wv = tid >> 5;
  const float* Fg = F + (size_t)b * STATE * STATE;
  const float* Pg = P + (size_t)b * STATE * STATE;
  for (int i = tid; i < STATE * STATE; i += 128) {
    Fs[i >> 6][i & 63] = (_Float16)Fg[i];
    Ps[i >> 6][i & 63] = (_Float16)Pg[i];
  }
  __syncthreads();
  { // Phase 1: M1 = F @ P   (P row-major -> K-major B)
    v8f acc[4] = {};
#pragma unroll
    for (int kb = 0; kb < 2; ++kb) {
      v16h a = frag_a(&Fs[0][0], STATE, wv * 16, kb * 32, lane);
#pragma unroll
      for (int j = 0; j < 4; ++j) {
        v16h bf = frag_b_kmajor(&Ps[0][0], STATE, j * 16, kb * 32, lane);
        acc[j] = WMMA_F16(a, bf, acc[j]);
      }
    }
    const int n = lane & 15, mg = (lane >> 4) * 8;
#pragma unroll
    for (int j = 0; j < 4; ++j)
#pragma unroll
      for (int r = 0; r < 8; ++r)
        M1[wv * 16 + mg + r][j * 16 + n] = (_Float16)acc[j][r];
  }
  __syncthreads();
  { // Phase 2: Pp = M1 @ F^T + diag(Q)  (F^T elem (k,n)=F[n][k] -> N-major B)
    v8f acc[4] = {};
#pragma unroll
    for (int kb = 0; kb < 2; ++kb) {
      v16h a = frag_a(&M1[0][0], STATE, wv * 16, kb * 32, lane);
#pragma unroll
      for (int j = 0; j < 4; ++j) {
        v16h bf = frag_b_nmajor(&Fs[0][0], STATE, j * 16, kb * 32, lane);
        acc[j] = WMMA_F16(a, bf, acc[j]);
      }
    }
    const int n = lane & 15, mg = (lane >> 4) * 8;
    float* Pout = Pp + (size_t)b * STATE * STATE;
    const float* Qb = Qd + (size_t)b * STATE;
#pragma unroll
    for (int j = 0; j < 4; ++j)
#pragma unroll
      for (int r = 0; r < 8; ++r) {
        const int row = wv * 16 + mg + r, col = j * 16 + n;
        float v = acc[j][r];
        if (row == col) v += Qb[row];
        Pout[row * STATE + col] = v;
      }
  }
}

// ------------------------- Tobit measurement update -------------------------
// One 128-thread block per batch: erf/exp statistics, 32x32 Gauss-Jordan
// inverse in LDS, state + covariance update. ~6% of FLOPs -> f32 VALU.
__global__ __launch_bounds__(128) void tobit_kernel(
    const float* __restrict__ xt, const float* __restrict__ Hg,
    const float* __restrict__ Tl, const float* __restrict__ Tu,
    const float* __restrict__ Rdg, const float* __restrict__ Ppred,
    float* __restrict__ y, float* __restrict__ Pout, float* __restrict__ out2) {
  __shared__ float Ps[STATE][STATE];
  __shared__ float Hs[OBS][STATE];
  __shared__ float Ys[STATE];
  __shared__ float pA[OBS], dyA[OBS], RnA[OBS], colk[OBS];
  __shared__ float R1s[STATE][OBS];
  __shared__ float R2s[OBS][2 * OBS];
  __shared__ float Ks[STATE][OBS];
  __shared__ float W2[OBS][STATE];
  const int b = blockIdx.x, tid = threadIdx.x;

  for (int i = tid; i < STATE * STATE; i += 128) Ps[i >> 6][i & 63] = Ppred[(size_t)b * STATE * STATE + i];
  for (int i = tid; i < OBS * STATE; i += 128)   Hs[i >> 6][i & 63] = Hg[i];
  if (tid < STATE) Ys[tid] = y[(size_t)b * STATE + tid];
  __syncthreads();

  if (tid < OBS) {
    const int j = tid;
    float Rdv = Rdg[(size_t)b * OBS + j];
    float r = sqrtf(Rdv) + EPSF;
    float z = 0.f;
    for (int k = 0; k < STATE; ++k) z += Hs[j][k] * Ys[k];
    float zl = (Tl[j] - z) / r, zu = (Tu[j] - z) / r;
    float cpl = ncdf_(zl), cpu_ = ncdf_(zu);
    float ppl = npdf_(zl), ppu = npdf_(zu);
    float p = cpu_ - cpl + EPSF;
    float lam = (ppu - ppl) / p;
    float cc = zl * ppl - zu * ppu;
    float Ey = p * (z - r * lam) + cpl * Tl[j] + (1.f - cpu_) * Tu[j];
    pA[j] = p;
    RnA[j] = Rdv * (cc / p + 1.f - lam * lam);
    dyA[j] = xt[(size_t)b * OBS + j] - Ey;
  }
  __syncthreads();

  // R1 = (P H^T) .* p[col]
  for (int e = tid; e < STATE * OBS; e += 128) {
    const int i = e >> 5, j = e & 31;
    float s = 0.f;
    for (int k = 0; k < STATE; ++k) s += Ps[i][k] * Hs[j][k];
    R1s[i][j] = s * pA[j];
  }
  __syncthreads();

  // R2 = p[row] .* (H R1) + diag(Rnew), augmented with identity
  for (int e = tid; e < OBS * OBS; e += 128) {
    const int i = e >> 5, j = e & 31;
    float s = 0.f;
    for (int k = 0; k < STATE; ++k) s += Hs[i][k] * R1s[k][j];
    s *= pA[i];
    if (i == j) s += RnA[i];
    R2s[i][j] = s;
    R2s[i][OBS + j] = (i == j) ? 1.f : 0.f;
  }
  __syncthreads();

  // Gauss-Jordan inverse of R2 (diagonally dominated; no pivoting)
  for (int k = 0; k < OBS; ++k) {
    const float piv = R2s[k][k];
    __syncthreads();
    if (tid < 2 * OBS) R2s[k][tid] = R2s[k][tid] / piv;
    __syncthreads();
    if (tid < OBS) colk[tid] = (tid == k) ? 0.f : R2s[tid][k];
    __syncthreads();
    for (int e = tid; e < OBS * 2 * OBS; e += 128) {
      const int i = e >> 6, cj = e & 63;
      if (i != k) R2s[i][cj] -= colk[i] * R2s[k][cj];
    }
    __syncthreads();
  }

  // K = R1 inv(R2)
  for (int e = tid; e < STATE * OBS; e += 128) {
    const int i = e >> 5, j = e & 31;
    float s = 0.f;
    for (int k = 0; k < OBS; ++k) s += R1s[i][k] * R2s[k][OBS + j];
    Ks[i][j] = s;
  }
  // W2 = p[row] .* (H P)
  for (int e = tid; e < OBS * STATE; e += 128) {
    const int j = e >> 6, l = e & 63;
    float s = 0.f;
    for (int k = 0; k < STATE; ++k) s += Hs[j][k] * Ps[k][l];
    W2[j][l] = pA[j] * s;
  }
  __syncthreads();

  if (tid < STATE) {
    float s = Ys[tid];
    for (int j = 0; j < OBS; ++j) s += Ks[tid][j] * dyA[j];
    y[(size_t)b * STATE + tid] = s;
    out2[(size_t)b * STATE + tid] = s;
  }
  for (int e = tid; e < STATE * STATE; e += 128) {
    const int i = e >> 6, l = e & 63;
    float s = 0.f;
    for (int j = 0; j < OBS; ++j) s += Ks[i][j] * W2[j][l];
    Pout[(size_t)b * STATE * STATE + e] = Ps[i][l] - s;
  }
}

// ------------------------------- host driver --------------------------------
extern "C" void kernel_launch(void* const* d_in, const int* in_sizes, int n_in,
                              void* d_out, int out_size, void* d_ws, size_t ws_size,
                              hipStream_t stream) {
  (void)n_in; (void)out_size; (void)ws_size;
  const float* x  = (const float*)d_in[0];
  const float* y0 = (const float*)d_in[1];
  const float* P0 = (const float*)d_in[2];
  const float* H  = (const float*)d_in[3];
  const float* Tl = (const float*)d_in[4];
  const float* Tu = (const float*)d_in[5];
  // params flattened in dict order: {cell{wih,whh,bih,bhh}, fc{w,b}} x4
  const float *wih[4], *whh[4], *bih[4], *bhh[4], *fw[4], *fb[4];
  for (int c = 0; c < 4; ++c) {
    const int base = 6 + c * 6;
    wih[c] = (const float*)d_in[base + 0];
    whh[c] = (const float*)d_in[base + 1];
    bih[c] = (const float*)d_in[base + 2];
    bhh[c] = (const float*)d_in[base + 3];
    fw[c]  = (const float*)d_in[base + 4];
    fb[c]  = (const float*)d_in[base + 5];
  }
  const int fcN[4] = {STATE, STATE * STATE, STATE, OBS};
  const int T = in_sizes[0] / (B_SZ * OBS);

  // workspace carve (256B aligned)
  char* wsb = (char*)d_ws;
  size_t off = 0;
  auto alloc = [&](size_t bytes) -> char* {
    size_t s = (off + 255) & ~size_t(255);
    off = s + bytes;
    return wsb + s;
  };
  _Float16* wcat[4]; float* bias[4]; _Float16* fw16[4];
  for (int c = 0; c < 4; ++c) wcat[c] = (_Float16*)alloc((size_t)GATES * KCAT * 2);
  for (int c = 0; c < 4; ++c) bias[c] = (float*)alloc((size_t)GATES * 4);
  for (int c = 0; c < 4; ++c) fw16[c] = (_Float16*)alloc((size_t)fcN[c] * HID * 2);
  float* hbuf[4]; float* cbuf[4];
  for (int c = 0; c < 4; ++c) {
    hbuf[c] = (float*)alloc((size_t)B_SZ * HID * 4);
    cbuf[c] = (float*)alloc((size_t)B_SZ * HID * 4);
  }
  float* yst   = (float*)alloc((size_t)B_SZ * STATE * 4);
  float* Pbuf  = (float*)alloc((size_t)B_SZ * STATE * STATE * 4);
  float* Ppbuf = (float*)alloc((size_t)B_SZ * STATE * STATE * 4);
  float* gates = (float*)alloc((size_t)B_SZ * GATES * 4);
  float* Fbuf  = (float*)alloc((size_t)B_SZ * STATE * STATE * 4);
  float* Qd    = (float*)alloc((size_t)B_SZ * STATE * 4);
  float* Rd    = (float*)alloc((size_t)B_SZ * OBS * 4);

  float* out1 = (float*)d_out;
  float* out2 = out1 + (size_t)T * B_SZ * STATE;

  // ---- prep: pack weights to f16, combine biases, init state ----
  for (int c = 0; c < 4; ++c) {
    pack_lstm_w_kernel<<<(GATES * KCAT + 255) / 256, 256, 0, stream>>>(wih[c], whh[c], wcat[c]);
    combine_bias_kernel<<<(GATES + 255) / 256, 256, 0, stream>>>(bih[c], bhh[c], bias[c]);
    cvt_f16_kernel<<<(fcN[c] * HID + 255) / 256, 256, 0, stream>>>(fw[c], fw16[c], fcN[c] * HID);
    zero_kernel<<<(B_SZ * HID + 255) / 256, 256, 0, stream>>>(hbuf[c], B_SZ * HID);
    zero_kernel<<<(B_SZ * HID + 255) / 256, 256, 0, stream>>>(cbuf[c], B_SZ * HID);
  }
  bcast_y_kernel<<<(B_SZ * STATE + 255) / 256, 256, 0, stream>>>(y0, yst);
  bcast_P_kernel<<<(B_SZ * STATE * STATE + 255) / 256, 256, 0, stream>>>(P0, Pbuf);

  // ---- sequential time loop ----
  const dim3 gGates(GATES / 128, B_SZ / 64);
  const dim3 gFfc((STATE * STATE) / 128, B_SZ / 64);
  for (int t = 0; t < T; ++t) {
    const float* xt = x + (size_t)t * B_SZ * OBS;
    // mean LSTM (input = previous posterior y) -> mean_fc -> prior y
    big_gemm_kernel<<<gGates, 256, 0, stream>>>(yst, hbuf[0], wcat[0], nullptr,
                                                gates, GATES, KCAT);
    lstm_pointwise_kernel<<<(B_SZ * HID + 255) / 256, 256, 0, stream>>>(gates, bias[0], hbuf[0], cbuf[0]);
    fc_gemm_kernel<<<dim3(STATE / 32, B_SZ / 64), 256, 0, stream>>>(
        hbuf[0], fw16[0], fb[0], yst, out1 + (size_t)t * B_SZ * STATE, STATE, 0);
    // F / Q / R LSTM cells on the new prior y
    for (int c = 1; c < 4; ++c) {
      big_gemm_kernel<<<gGates, 256, 0, stream>>>(yst, hbuf[c], wcat[c], nullptr,
                                                  gates, GATES, KCAT);
      lstm_pointwise_kernel<<<(B_SZ * HID + 255) / 256, 256, 0, stream>>>(gates, bias[c], hbuf[c], cbuf[c]);
    }
    big_gemm_kernel<<<gFfc, 256, 0, stream>>>(nullptr, hbuf[1], fw16[1], fb[1],
                                              Fbuf, STATE * STATE, HID);
    fc_gemm_kernel<<<dim3(STATE / 32, B_SZ / 64), 256, 0, stream>>>(
        hbuf[2], fw16[2], fb[2], Qd, nullptr, STATE, 1);
    fc_gemm_kernel<<<dim3(OBS / 32, B_SZ / 64), 256, 0, stream>>>(
        hbuf[3], fw16[3], fb[3], Rd, nullptr, OBS, 1);
    // covariance prediction + Tobit measurement update
    ppred_kernel<<<B_SZ, 128, 0, stream>>>(Fbuf, Pbuf, Qd, Ppbuf);
    tobit_kernel<<<B_SZ, 128, 0, stream>>>(xt, H, Tl, Tu, Rd, Ppbuf, yst, Pbuf,
                                           out2 + (size_t)t * B_SZ * STATE);
  }
}
